// MoE_52243982188859
// MI455X (gfx1250) — matrix operand order repaired and checked by reading, not compiled, not run on hardware
//
#include <hip/hip_runtime.h>
#include <hip/hip_bf16.h>
#include <math.h>

// ---------------------------------------------------------------------------
// MoE (dense experts, top-2 gated combine) for MI455X / gfx1250, wave32 WMMA.
//   Pass 0: gating  -> weights[T][E] (f32, softmax probs masked to top-2)
//   Pass 1: H'[e,t,f] = gate[t,e] * gelu(x·w1_e^T) * (x·w3_e^T)   (bf16)
//           wave tile 32M x 32N dual (8 acc frags = 64 VGPRs, spill-free)
//   Pass 2: out[t,d] = sum_e H'_e · w2_e^T                        (bf16)
//           wave tile 64M x 32N (8 acc frags)
// Both GEMM passes are K-contiguous on A and B, so WMMA fragments are
// direct b128 global loads (no LDS transpose). f32 accumulation via
// v_wmma_f32_16x16x32_bf16. Accumulator footprint capped at 64 VGPRs to
// stay inside the 256-VGPR direct window (round 2's 128-VGPR acc spilled).
// Workspace: [0,16KB) gate weights, then 64MB H'.
// ---------------------------------------------------------------------------

typedef __bf16 bf16;
typedef bf16  v8bf  __attribute__((ext_vector_type(8)));
typedef bf16  v16bf __attribute__((ext_vector_type(16)));
typedef float v8f   __attribute__((ext_vector_type(8)));

#define D_DIM 8192
#define F_DIM 16384
#define E_NUM 4
#define T_NUM 512

__device__ inline v8f zero8() {
    v8f z;
    #pragma unroll
    for (int i = 0; i < 8; ++i) z[i] = 0.0f;
    return z;
}

// A fragment: 16x32 bf16, rows m0..m0+15, cols k0..k0+31, row-major, ld = row len.
// ISA layout: lanes 0-15 hold K = k0+[0,8) and k0+[16,24); lanes 16-31 hold
// K = k0+[8,16) and k0+[24,32); row M = lane & 15.
__device__ inline v16bf load_a_frag(const bf16* __restrict__ base, int ld,
                                    int m0, int k0, int lr, int half) {
    const bf16* row = base + (size_t)(m0 + lr) * (size_t)ld;
    v8bf lo = *(const v8bf*)(row + k0 + half * 8);
    v8bf hi = *(const v8bf*)(row + k0 + 16 + half * 8);
    return __builtin_shufflevector(lo, hi, 0,1,2,3,4,5,6,7,8,9,10,11,12,13,14,15);
}

// B fragment: 32x16 bf16 (KxN). Source is row-major W with N = weight-row
// index and K contiguous along the row (NT GEMM). Lanes 0-15: K = k0+[0,8) and
// k0+[16,24), lanes 16-31: K = k0+[8,16) and k0+[24,32); column N = lane & 15.
__device__ inline v16bf load_b_frag(const bf16* __restrict__ base, int ld,
                                    int n0, int k0, int lr, int half) {
    const bf16* row = base + (size_t)(n0 + lr) * (size_t)ld;
    const bf16* p = row + k0 + half * 8;
    v8bf lo = *(const v8bf*)(p);
    v8bf hi = *(const v8bf*)(p + 16);
    return __builtin_shufflevector(lo, hi, 0,1,2,3,4,5,6,7,8,9,10,11,12,13,14,15);
}

__device__ inline v8f wmma_bf16(v16bf a, v16bf b, v8f c) {
    return __builtin_amdgcn_wmma_f32_16x16x32_bf16(
        /*neg_a=*/false, a, /*neg_b=*/false, b,
        /*c_mod=*/(short)0, c, /*reuse_a=*/false, /*reuse_b=*/false);
}

// ---------------- Pass 0: top-2 softmax gating (one wave per token) --------
__global__ void moe_gate_kernel(const bf16* __restrict__ x,
                                const bf16* __restrict__ gate_w,
                                float* __restrict__ weights) {
    const int t = blockIdx.x;
    const int lane = threadIdx.x & 31;
    const bf16* xr = x + (size_t)t * D_DIM;

    float s[E_NUM] = {0.f, 0.f, 0.f, 0.f};
    for (int d = lane * 8; d < D_DIM; d += 32 * 8) {
        v8bf xv = *(const v8bf*)(xr + d);
        #pragma unroll
        for (int e = 0; e < E_NUM; ++e) {
            v8bf gv = *(const v8bf*)(gate_w + (size_t)e * D_DIM + d);
            #pragma unroll
            for (int i = 0; i < 8; ++i)
                s[e] += (float)xv[i] * (float)gv[i];
        }
    }
    #pragma unroll
    for (int off = 16; off > 0; off >>= 1)
        #pragma unroll
        for (int e = 0; e < E_NUM; ++e)
            s[e] += __shfl_xor(s[e], off, 32);

    if (lane == 0) {
        float m = fmaxf(fmaxf(s[0], s[1]), fmaxf(s[2], s[3]));
        float p[E_NUM], sum = 0.f;
        #pragma unroll
        for (int e = 0; e < E_NUM; ++e) { p[e] = __expf(s[e] - m); sum += p[e]; }
        float inv = 1.0f / sum;
        #pragma unroll
        for (int e = 0; e < E_NUM; ++e) p[e] *= inv;
        int i0 = 0;
        #pragma unroll
        for (int e = 1; e < E_NUM; ++e) if (p[e] > p[i0]) i0 = e;
        int i1 = (i0 == 0) ? 1 : 0;
        #pragma unroll
        for (int e = 0; e < E_NUM; ++e)
            if (e != i0 && p[e] > p[i1]) i1 = e;
        #pragma unroll
        for (int e = 0; e < E_NUM; ++e)
            weights[t * E_NUM + e] = (e == i0 || e == i1) ? p[e] : 0.0f;
    }
}

// ---------------- Pass 1: H' = gate * gelu(X W1^T) * (X W3^T) --------------
// Wave tile: M=32 (tokens) x N=32 (hidden f), dual accumulators for w1/w3.
// Per K-step: 2 A frags (4 loads, reused x4 each), 4 B frags (8 loads)
// -> 8 WMMA / 12 b128 loads. 8 acc frags = 64 VGPRs (spill-free).
// Grid: (F/32, T/32, E), block = 1 wave (32 threads).
__global__ void moe_h_kernel(const bf16* __restrict__ x,    // (T, D)
                             const bf16* __restrict__ w1,   // (E, F, D)
                             const bf16* __restrict__ w3,   // (E, F, D)
                             const float* __restrict__ gw,  // (T, E)
                             bf16* __restrict__ H) {        // (E, T, F)
    const int lane = threadIdx.x & 31;
    const int lr = lane & 15, half = lane >> 4;
    const int n0 = blockIdx.x * 32;
    const int m0 = blockIdx.y * 32;
    const int e  = blockIdx.z;

    const bf16* w1e = w1 + (size_t)e * F_DIM * D_DIM;
    const bf16* w3e = w3 + (size_t)e * F_DIM * D_DIM;

    v8f acc1[2][2], acc3[2][2];
    #pragma unroll
    for (int mi = 0; mi < 2; ++mi)
        #pragma unroll
        for (int n = 0; n < 2; ++n) { acc1[mi][n] = zero8(); acc3[mi][n] = zero8(); }

    for (int k0 = 0; k0 < D_DIM; k0 += 32) {
        v16bf a0 = load_a_frag(x, D_DIM, m0,      k0, lr, half);
        v16bf a1 = load_a_frag(x, D_DIM, m0 + 16, k0, lr, half);
        #pragma unroll
        for (int n = 0; n < 2; ++n) {
            v16bf b1 = load_b_frag(w1e, D_DIM, n0 + 16 * n, k0, lr, half);
            acc1[0][n] = wmma_bf16(a0, b1, acc1[0][n]);
            acc1[1][n] = wmma_bf16(a1, b1, acc1[1][n]);
            v16bf b3 = load_b_frag(w3e, D_DIM, n0 + 16 * n, k0, lr, half);
            acc3[0][n] = wmma_bf16(a0, b3, acc3[0][n]);
            acc3[1][n] = wmma_bf16(a1, b3, acc3[1][n]);
        }
    }

    // Epilogue: exact-erf GELU, gate-weight fold, bf16 store.
    // C/D layout: element i of the v8f lives at row M = i + 8*half, col N = lr.
    bf16* Hb = H + (size_t)e * T_NUM * F_DIM;
    #pragma unroll
    for (int mi = 0; mi < 2; ++mi) {
        float g8[8];
        #pragma unroll
        for (int i = 0; i < 8; ++i)
            g8[i] = gw[(m0 + 16 * mi + i + 8 * half) * E_NUM + e];
        #pragma unroll
        for (int n = 0; n < 2; ++n) {
            #pragma unroll
            for (int i = 0; i < 8; ++i) {
                float h1 = acc1[mi][n][i];
                float h3 = acc3[mi][n][i];
                float gelu = 0.5f * h1 * (1.0f + erff(h1 * 0.70710678118654752f));
                float hv = gelu * h3 * g8[i];
                Hb[(size_t)(m0 + 16 * mi + i + 8 * half) * F_DIM +
                   (n0 + 16 * n + lr)] = (bf16)hv;
            }
        }
    }
}

// ---------------- Pass 2: out = sum_e H'_e W2_e^T --------------------------
// Wave tile: M=64 (tokens) x N=32 (model dim d). 8 acc frags = 64 VGPRs.
// Per K-step: 4 A frags (8 loads), 2 B frags (4 loads) -> 8 WMMA / 12 loads.
// Grid: (D/32, T/64).
__global__ void moe_out_kernel(const bf16* __restrict__ H,   // (E, T, F)
                               const bf16* __restrict__ w2,  // (E, D, F)
                               bf16* __restrict__ out) {     // (T, D)
    const int lane = threadIdx.x & 31;
    const int lr = lane & 15, half = lane >> 4;
    const int n0 = blockIdx.x * 32;
    const int m0 = blockIdx.y * 64;

    v8f acc[4][2];
    #pragma unroll
    for (int mi = 0; mi < 4; ++mi)
        #pragma unroll
        for (int n = 0; n < 2; ++n) acc[mi][n] = zero8();

    for (int e = 0; e < E_NUM; ++e) {
        const bf16* He  = H  + (size_t)e * T_NUM * F_DIM;
        const bf16* w2e = w2 + (size_t)e * D_DIM * F_DIM;
        for (int k0 = 0; k0 < F_DIM; k0 += 32) {
            v16bf a[4];
            #pragma unroll
            for (int mi = 0; mi < 4; ++mi)
                a[mi] = load_a_frag(He, F_DIM, m0 + 16 * mi, k0, lr, half);
            #pragma unroll
            for (int n = 0; n < 2; ++n) {
                v16bf b = load_b_frag(w2e, F_DIM, n0 + 16 * n, k0, lr, half);
                #pragma unroll
                for (int mi = 0; mi < 4; ++mi)
                    acc[mi][n] = wmma_bf16(a[mi], b, acc[mi][n]);
            }
        }
    }

    #pragma unroll
    for (int mi = 0; mi < 4; ++mi)
        #pragma unroll
        for (int n = 0; n < 2; ++n)
            #pragma unroll
            for (int i = 0; i < 8; ++i)
                out[(size_t)(m0 + 16 * mi + i + 8 * half) * D_DIM +
                    (n0 + 16 * n + lr)] = (bf16)acc[mi][n][i];
}

// ---------------------------------------------------------------------------
extern "C" void kernel_launch(void* const* d_in, const int* in_sizes, int n_in,
                              void* d_out, int out_size, void* d_ws, size_t ws_size,
                              hipStream_t stream) {
    const bf16* x      = (const bf16*)d_in[0];   // (1,1,T,D)
    const bf16* gate_w = (const bf16*)d_in[1];   // (E,D)
    const bf16* w1     = (const bf16*)d_in[2];   // (E,F,D)
    const bf16* w2     = (const bf16*)d_in[3];   // (E,D,F)
    const bf16* w3     = (const bf16*)d_in[4];   // (E,F,D)
    bf16* out = (bf16*)d_out;

    // Workspace: gate weights (T*E f32 = 8KB, padded to 16KB) then H' (64MB).
    float* weights = (float*)d_ws;
    bf16*  Hbuf    = (bf16*)((char*)d_ws + 16 * 1024);

    moe_gate_kernel<<<dim3(T_NUM), dim3(32), 0, stream>>>(x, gate_w, weights);

    dim3 gridH(F_DIM / 32, T_NUM / 32, E_NUM);
    moe_h_kernel<<<gridH, dim3(32), 0, stream>>>(x, w1, w3, weights, Hbuf);

    dim3 gridO(D_DIM / 32, T_NUM / 64, 1);
    moe_out_kernel<<<gridO, dim3(32), 0, stream>>>(Hbuf, w2, out);
}
